// M11_74466142978434
// MI455X (gfx1250) — compile-verified, matrix operand order checked by reading, not run on hardware
//
#include <hip/hip_runtime.h>

// GINEConv GNN (3 layers, DenseNet concat) + MLP head for MI455X (gfx1250).
// Dense node GEMMs run on v_wmma_f32_16x16x32_f16 (fp16 in / fp32 acc),
// register-blocked 16x64 per wave (4 accumulators reuse one A fragment ->
// 2.5 b128 loads per WMMA instead of 4). Edge message passing uses
// L2-resident global_atomic_add_f32 scatter (node tables fit the 192MB L2).
//
// Problem sizes: M=50000=3125*16, Nc in {128,512} (mult of 64),
// K in {64,128,192,320,448,512} (mult of 32)  => no WMMA edge masking needed.

#define NN   50000
#define NE   800000
#define FNODE 64
#define FEDGE 16
#define CONV 128
#define MLPC 512
#define HTOT 448  // 64 + 3*128

typedef __attribute__((ext_vector_type(16))) _Float16 v16h;
typedef __attribute__((ext_vector_type(8)))  _Float16 v8h;
typedef __attribute__((ext_vector_type(8)))  float    v8f;

// ---------------------------------------------------------------- utilities
__global__ void zero_f32(float* __restrict__ p, long n) {
  long stride = (long)gridDim.x * blockDim.x;
  for (long i = (long)blockIdx.x * blockDim.x + threadIdx.x; i < n; i += stride)
    p[i] = 0.0f;
}

__global__ void copy_cols(const float* __restrict__ src, int lds,
                          float* __restrict__ dst, int ldd, int N, int C) {
  long total = (long)N * C;
  long stride = (long)gridDim.x * blockDim.x;
  for (long i = (long)blockIdx.x * blockDim.x + threadIdx.x; i < total; i += stride) {
    int n = (int)(i / C), c = (int)(i % C);
    dst[(size_t)n * ldd + c] = src[(size_t)n * lds + c];
  }
}

__global__ void cvt_f32_f16(const float* __restrict__ s, _Float16* __restrict__ d, long n) {
  long stride = (long)gridDim.x * blockDim.x;
  for (long i = (long)blockIdx.x * blockDim.x + threadIdx.x; i < n; i += stride)
    d[i] = (_Float16)s[i];
}

// --------------------------------------------------- batch-norm statistics
// one block per channel; reduce mean / biased var over N rows (row stride ldX)
__global__ __launch_bounds__(256)
void col_stats(const float* __restrict__ X, int ldX, int N,
               float* __restrict__ mean, float* __restrict__ rstd) {
  __shared__ float s1[256], s2[256];
  const int c = blockIdx.x;
  float sum = 0.0f, sq = 0.0f;
  for (int n = threadIdx.x; n < N; n += 256) {
    float v = X[(size_t)n * ldX + c];
    sum += v; sq += v * v;
  }
  s1[threadIdx.x] = sum; s2[threadIdx.x] = sq;
  __syncthreads();
  for (int off = 128; off > 0; off >>= 1) {
    if ((int)threadIdx.x < off) {
      s1[threadIdx.x] += s1[threadIdx.x + off];
      s2[threadIdx.x] += s2[threadIdx.x + off];
    }
    __syncthreads();
  }
  if (threadIdx.x == 0) {
    float m = s1[0] / (float)N;
    float var = s2[0] / (float)N - m * m;
    mean[c] = m;
    rstd[c] = rsqrtf(var + 1e-5f);
  }
}

__global__ void bn_act(const float* __restrict__ X, int ldX,
                       float* __restrict__ Y, int ldY,
                       const float* __restrict__ mean, const float* __restrict__ rstd,
                       const float* __restrict__ g, const float* __restrict__ b,
                       int N, int C, int doLrelu) {
  long total = (long)N * C;
  long stride = (long)gridDim.x * blockDim.x;
  for (long i = (long)blockIdx.x * blockDim.x + threadIdx.x; i < total; i += stride) {
    int n = (int)(i / C), c = (int)(i % C);
    float v = (X[(size_t)n * ldX + c] - mean[c]) * rstd[c] * g[c] + b[c];
    if (doLrelu) v = v > 0.0f ? v : 0.01f * v;
    Y[(size_t)n * ldY + c] = v;
  }
}

// -------------------------------------------- fused edge linear + gather +
// relu + scatter-sum (atomics resolve in L2; node table is L2 resident)
__global__ __launch_bounds__(256)
void edge_msg(const float* __restrict__ hin, const int* __restrict__ srcI,
              const int* __restrict__ dstI, const float* __restrict__ ea,
              const float* __restrict__ We, const float* __restrict__ be,
              float* __restrict__ aggr, int E, int C) {
  extern __shared__ float sW[];  // C*16 weights + C bias
  for (int i = threadIdx.x; i < C * 17; i += blockDim.x)
    sW[i] = (i < C * 16) ? We[i] : be[i - C * 16];
  __syncthreads();
  long total = (long)E * C;
  long stride = (long)gridDim.x * blockDim.x;
  for (long idx = (long)blockIdx.x * blockDim.x + threadIdx.x; idx < total; idx += stride) {
    int e = (int)(idx / C), c = (int)(idx % C);
    const float* eap = ea + (size_t)e * FEDGE;
    const float* wp = sW + c * FEDGE;
    float acc = sW[C * FEDGE + c];
#pragma unroll
    for (int k = 0; k < FEDGE; ++k) acc += eap[k] * wp[k];
    float m = hin[(size_t)srcI[e] * C + c] + acc;
    m = m > 0.0f ? m : 0.0f;
    atomicAdd(&aggr[(size_t)dstI[e] * C + c], m);
  }
}

// z = (1+eps)*hin + aggr   (z may alias aggr)
__global__ void gine_combine(const float* __restrict__ hin, const float* aggr,
                             const float* __restrict__ eps, float* z, long n) {
  float s = 1.0f + eps[0];
  long stride = (long)gridDim.x * blockDim.x;
  for (long i = (long)blockIdx.x * blockDim.x + threadIdx.x; i < n; i += stride)
    z[i] = s * hin[i] + aggr[i];
}

// ------------------------------------------------------------ WMMA GEMM
// out[M,Nc](ldOut) = A[M,K] @ W[Nc,K]^T + bias.
// One wave computes a 16(M) x 64(N) block: 4 f32 accumulators share one A
// fragment per 32-wide k-step (5 fragment loads : 4 WMMAs).
// Per-lane fragment layout per CDNA5 ISA 7.12.2 (16-bit A 16x32): lane = row
// (or col) mod 16; lane>>4 selects K-subrange; element e ->
// k = 16*(e/8) + 8*(lane>>4) + e%8  => two 16B v8h loads per fragment.
__device__ __forceinline__ v16h ld_frag(const _Float16* p) {
  v8h lo = *(const v8h*)(p);
  v8h hi = *(const v8h*)(p + 16);
  return __builtin_shufflevector(lo, hi, 0, 1, 2, 3, 4, 5, 6, 7,
                                 8, 9, 10, 11, 12, 13, 14, 15);
}
__device__ __forceinline__ v8f wmma_f16(v16h a, v16h b, v8f c) {
  return __builtin_amdgcn_wmma_f32_16x16x32_f16(false, a, false, b,
                                                (short)0, c, false, false);
}

__global__ __launch_bounds__(256)
void wmma_gemm4(const _Float16* __restrict__ A, const _Float16* __restrict__ W,
                const float* __restrict__ bias, float* __restrict__ out,
                int M, int K, int Nc, int ldOut) {
  const int lane = threadIdx.x & 31;
  const long wave = (long)blockIdx.x * (blockDim.x >> 5) + (threadIdx.x >> 5);
  const int nBlk = Nc >> 6;  // 64-wide N blocks
  const long totalBlks = (long)(M >> 4) * nBlk;
  if (wave >= totalBlks) return;  // wave-uniform; EXEC all-ones at WMMA
  const int tM = (int)(wave / nBlk);
  const int tN = (int)(wave % nBlk);
  const int half = lane >> 4;
  const int l = lane & 15;
  const _Float16* arow = A + (size_t)(tM * 16 + l) * K;
  const _Float16* wrow = W + (size_t)(tN * 64 + l) * K;
  const size_t wstep = (size_t)16 * K;  // 16 output columns
  v8f acc0 = {}, acc1 = {}, acc2 = {}, acc3 = {};
  for (int k0 = 0; k0 < K; k0 += 32) {
    const int ka = k0 + 8 * half;
    v16h a  = ld_frag(arow + ka);
    v16h b0 = ld_frag(wrow + ka);
    v16h b1 = ld_frag(wrow + wstep + ka);
    v16h b2 = ld_frag(wrow + 2 * wstep + ka);
    v16h b3 = ld_frag(wrow + 3 * wstep + ka);
    acc0 = wmma_f16(a, b0, acc0);
    acc1 = wmma_f16(a, b1, acc1);
    acc2 = wmma_f16(a, b2, acc2);
    acc3 = wmma_f16(a, b3, acc3);
  }
  // C/D layout: VGPR r -> row r + 8*(lane>>4); col = lane mod 16
  const int colBase = tN * 64 + l;
  float* orow = out + (size_t)(tM * 16 + 8 * half) * ldOut;
#pragma unroll
  for (int j = 0; j < 4; ++j) {
    const int col = colBase + j * 16;
    const float bv = bias ? bias[col] : 0.0f;
    const v8f acc = (j == 0) ? acc0 : (j == 1) ? acc1 : (j == 2) ? acc2 : acc3;
#pragma unroll
    for (int r = 0; r < 8; ++r)
      orow[(size_t)r * ldOut + col] = acc[r] + bv;
  }
}

// final 512->1 projection
__global__ void final_dot(const float* __restrict__ Y, const float* __restrict__ W3,
                          const float* __restrict__ b3, float* __restrict__ out,
                          int N, int K) {
  int n = blockIdx.x * blockDim.x + threadIdx.x;
  if (n >= N) return;
  const float* y = Y + (size_t)n * K;
  float acc = 0.0f;
  for (int k = 0; k < K; k += 4)
    acc += y[k] * W3[k] + y[k + 1] * W3[k + 1] + y[k + 2] * W3[k + 2] + y[k + 3] * W3[k + 3];
  out[n] = acc + b3[0];
}

// ---------------------------------------------------------------- driver
static inline int blocksFor(long n) {
  long b = (n + 255) / 256;
  return (int)(b > 16384 ? 16384 : b);
}

static inline void launch_gemm(const _Float16* A, const _Float16* W, const float* bias,
                               float* out, int M, int K, int Nc, int ldOut,
                               hipStream_t stream) {
  long waves = (long)(M / 16) * (Nc / 64);
  wmma_gemm4<<<(int)((waves + 7) / 8), 256, 0, stream>>>(A, W, bias, out, M, K, Nc, ldOut);
}

extern "C" void kernel_launch(void* const* d_in, const int* in_sizes, int n_in,
                              void* d_out, int out_size, void* d_ws, size_t ws_size,
                              hipStream_t stream) {
  (void)in_sizes; (void)n_in; (void)out_size; (void)ws_size;

  const float* x = (const float*)d_in[0];
  const int* ei = (const int*)d_in[1];
  const int* srcI = ei;            // edge_index[0]
  const int* dstI = ei + NE;       // edge_index[1]
  const float* eattr = (const float*)d_in[2];

  struct Layer { const float *bn_g, *bn_b, *eps, *We, *be, *W1, *b1, *g2, *bb2, *W2, *b2; };
  Layer L[3];
  for (int i = 0; i < 3; ++i) {
    int s = 3 + i * 11;
    L[i].bn_g = (const float*)d_in[s + 0];
    L[i].bn_b = (const float*)d_in[s + 1];
    L[i].eps  = (const float*)d_in[s + 2];
    L[i].We   = (const float*)d_in[s + 3];
    L[i].be   = (const float*)d_in[s + 4];
    L[i].W1   = (const float*)d_in[s + 5];
    L[i].b1   = (const float*)d_in[s + 6];
    L[i].g2   = (const float*)d_in[s + 7];
    L[i].bb2  = (const float*)d_in[s + 8];
    L[i].W2   = (const float*)d_in[s + 9];
    L[i].b2   = (const float*)d_in[s + 10];
  }
  const int cb = 3 + 3 * 11;
  const float* cW1 = (const float*)d_in[cb + 0];
  const float* cb1 = (const float*)d_in[cb + 1];
  const float* cg1 = (const float*)d_in[cb + 2];
  const float* cbb1 = (const float*)d_in[cb + 3];
  const float* cW2 = (const float*)d_in[cb + 4];
  const float* cb2 = (const float*)d_in[cb + 5];
  const float* cg2 = (const float*)d_in[cb + 6];
  const float* cbb2 = (const float*)d_in[cb + 7];
  const float* cW3 = (const float*)d_in[cb + 8];
  const float* cb3 = (const float*)d_in[cb + 9];

  // workspace layout (16B-aligned offsets), ~347 MB total
  char* ws = (char*)d_ws;
  float*    H    = (float*)(ws + 0);            // N x 448  (concat features)
  float*    B1   = (float*)(ws + 89600000);     // N x 512
  float*    B2   = (float*)(ws + 192000000);    // N x 512
  _Float16* F16  = (_Float16*)(ws + 294400000); // N x 512 f16 activations
  _Float16* WF16 = (_Float16*)(ws + 345600000); // 512x512 f16 weights
  float*    mean = (float*)(ws + 346200000);    // 512
  float*    rstd = mean + 512;                  // 512

  // H[:, :64] = x
  copy_cols<<<blocksFor((long)NN * FNODE), 256, 0, stream>>>(x, FNODE, H, HTOT, NN, FNODE);

  int inC = FNODE;
  for (int i = 0; i < 3; ++i) {
    long nc = (long)NN * inC;
    // hn = BN(h[:, :inC]); hin = (i>0) ? lrelu(hn) : hn   -> B1
    col_stats<<<inC, 256, 0, stream>>>(H, HTOT, NN, mean, rstd);
    bn_act<<<blocksFor(nc), 256, 0, stream>>>(H, HTOT, B1, inC, mean, rstd,
                                              L[i].bn_g, L[i].bn_b, NN, inC, i > 0);
    // aggr = scatter_sum(relu(hin[src] + ea@We^T + be), dst)  -> B2
    zero_f32<<<blocksFor(nc), 256, 0, stream>>>(B2, nc);
    edge_msg<<<8192, 256, (size_t)(inC * 17) * sizeof(float), stream>>>(
        B1, srcI, dstI, eattr, L[i].We, L[i].be, B2, NE, inC);
    // z = (1+eps)*hin + aggr  (in place in B2)
    gine_combine<<<blocksFor(nc), 256, 0, stream>>>(B1, B2, L[i].eps, B2, nc);
    // z @ W1^T + b1 -> B1 [N,128]   (WMMA)
    cvt_f32_f16<<<blocksFor(nc), 256, 0, stream>>>(B2, F16, nc);
    cvt_f32_f16<<<blocksFor((long)CONV * inC), 256, 0, stream>>>(L[i].W1, WF16, (long)CONV * inC);
    launch_gemm(F16, WF16, L[i].b1, B1, NN, inC, CONV, CONV, stream);
    // lrelu(BN(.)) -> B2
    col_stats<<<CONV, 256, 0, stream>>>(B1, CONV, NN, mean, rstd);
    bn_act<<<blocksFor((long)NN * CONV), 256, 0, stream>>>(B1, CONV, B2, CONV, mean, rstd,
                                                           L[i].g2, L[i].bb2, NN, CONV, 1);
    // z2 = . @ W2^T + b2  written straight into the concat slot of H (WMMA)
    cvt_f32_f16<<<blocksFor((long)NN * CONV), 256, 0, stream>>>(B2, F16, (long)NN * CONV);
    cvt_f32_f16<<<blocksFor((long)CONV * CONV), 256, 0, stream>>>(L[i].W2, WF16, (long)CONV * CONV);
    launch_gemm(F16, WF16, L[i].b2, H + (FNODE + i * CONV), NN, CONV, CONV, HTOT, stream);
    inC = FNODE + (i + 1) * CONV;
  }

  // classifier: lrelu(BN(H@W1^T+b1)) -> lrelu(BN(.@W2^T+b2)) -> .@W3^T+b3
  cvt_f32_f16<<<blocksFor((long)NN * HTOT), 256, 0, stream>>>(H, F16, (long)NN * HTOT);
  cvt_f32_f16<<<blocksFor((long)MLPC * HTOT), 256, 0, stream>>>(cW1, WF16, (long)MLPC * HTOT);
  launch_gemm(F16, WF16, cb1, B1, NN, HTOT, MLPC, MLPC, stream);
  col_stats<<<MLPC, 256, 0, stream>>>(B1, MLPC, NN, mean, rstd);
  bn_act<<<blocksFor((long)NN * MLPC), 256, 0, stream>>>(B1, MLPC, B2, MLPC, mean, rstd,
                                                         cg1, cbb1, NN, MLPC, 1);
  cvt_f32_f16<<<blocksFor((long)NN * MLPC), 256, 0, stream>>>(B2, F16, (long)NN * MLPC);
  cvt_f32_f16<<<blocksFor((long)MLPC * MLPC), 256, 0, stream>>>(cW2, WF16, (long)MLPC * MLPC);
  launch_gemm(F16, WF16, cb2, B1, NN, MLPC, MLPC, MLPC, stream);
  col_stats<<<MLPC, 256, 0, stream>>>(B1, MLPC, NN, mean, rstd);
  bn_act<<<blocksFor((long)NN * MLPC), 256, 0, stream>>>(B1, MLPC, B2, MLPC, mean, rstd,
                                                         cg2, cbb2, NN, MLPC, 1);
  final_dot<<<(NN + 255) / 256, 256, 0, stream>>>(B2, cW3, cb3, (float*)d_out, NN, MLPC);
}
